// Node_91001767068023
// MI455X (gfx1250) — compile-verified, matrix operand order checked by reading, not run on hardware
//
#include <hip/hip_runtime.h>
#include <hip/hip_bf16.h>

// ---------------------------------------------------------------------------
// TOPK == NUM_CODES (16) => top_k selects ALL codes; mean(selected) is the
// constant mean of the codebook rows => projector/cdist/topk is dead code.
// Remaining work: two big GEMMs (AI ~ 500 FLOP/B, compute-bound), executed
// with split-bf16 (hi+lo, 3 products) WMMA for fp32-class accuracy at bf16
// matrix-engine throughput. h (128 MB) is L2-resident (192 MB L2).
// Tile: 128x128 per workgroup, 8 waves x (32x64), 24 WMMAs/wave/K-step.
// Global->register double buffering hides HBM latency under the WMMA phase.
// ---------------------------------------------------------------------------

typedef __bf16 v16bf __attribute__((ext_vector_type(16)));
typedef float  v8f   __attribute__((ext_vector_type(8)));

union FragU { v16bf v; uint4 q[2]; };
union Pack4 { __bf16 b[4]; uint2 u; };

#define BM 128
#define BN 128
#define BK 32
#define LDSROW 40   // 32 + 8 pad: stride 80B -> conflict-free b128 frag reads

// Load one 16x32 bf16 fragment from LDS, per the CDNA5 16-bit A-matrix VGPR
// layout (CDNA5 ISA):
//   lanes 0-15 : row = lane,    K = {0..7, 16..23}
//   lanes 16-31: row = lane-16, K = {8..15, 24..31}
__device__ __forceinline__ v16bf ldsFrag(const __bf16* base, int lane) {
    const int row  = lane & 15;
    const int half = (lane >> 4) & 1;
    const __bf16* p = base + row * LDSROW + half * 8;
    FragU u;
    u.q[0] = *reinterpret_cast<const uint4*>(p);
    u.q[1] = *reinterpret_cast<const uint4*>(p + 16);
    return u.v;
}

// b1' = b1 + W1[:, 2048:2560] @ mean(codebook, axis=0)
__global__ __launch_bounds__(1024) void prep_bias(
    const float* __restrict__ codebook,   // [16][512]
    const float* __restrict__ W1,         // [1024][2560]
    const float* __restrict__ b1,         // [1024]
    float* __restrict__ b1p)              // [1024]
{
    __shared__ float mc[512];
    const int t = threadIdx.x;
    if (t < 512) {
        float s = 0.0f;
        #pragma unroll
        for (int c = 0; c < 16; ++c) s += codebook[c * 512 + t];
        mc[t] = s * (1.0f / 16.0f);
    }
    __syncthreads();
    float acc = b1[t];
    const float* wrow = W1 + (size_t)t * 2560 + 2048;
    for (int s = 0; s < 512; ++s) acc += wrow[s] * mc[s];
    b1p[t] = acc;
}

// Split fp32 weights into bf16 hi + bf16 lo (residual).
__global__ void split_w(const float* __restrict__ src, int ld, int cols, int rows,
                        __bf16* __restrict__ hi, __bf16* __restrict__ lo)
{
    const int n = rows * cols;
    for (int i = blockIdx.x * blockDim.x + threadIdx.x; i < n;
         i += gridDim.x * blockDim.x) {
        const int r = i / cols, c = i - r * cols;
        const float v = src[(size_t)r * ld + c];
        const __bf16 h = (__bf16)v;
        hi[i] = h;
        lo[i] = (__bf16)(v - (float)h);
    }
}

// C[M,N] = act(A[M,K] @ B[N,K]^T + bias), A fp32 (split on the fly),
// B pre-split bf16 hi/lo. Workgroup tile 128x128, 8 waves of 32x64.
__global__ __launch_bounds__(256) void gemm_bf16x2(
    const float* __restrict__ A, int lda,
    const __bf16* __restrict__ Bhi, const __bf16* __restrict__ Blo, int ldb,
    const float* __restrict__ bias,
    float* __restrict__ C, int ldc,
    int K, int doRelu)
{
    __shared__ __bf16 sAhi[BM * LDSROW];
    __shared__ __bf16 sAlo[BM * LDSROW];
    __shared__ __bf16 sBhi[BN * LDSROW];
    __shared__ __bf16 sBlo[BN * LDSROW];

    const int tid  = threadIdx.x;
    const int lane = tid & 31;
    const int wave = tid >> 5;       // 0..7
    const int wm   = wave >> 1;      // 0..3 : 32-row slab
    const int wn   = wave & 1;       // 0..1 : 64-col slab

    const int m0 = blockIdx.x * BM;
    const int n0 = blockIdx.y * BN;

    // Per-thread staging coordinates: element e = tid*4 + i*1024
    //  -> row = r0 + 32*i, col = kc (constant).
    const int r0 = (tid * 4) >> 5;   // 0..31
    const int kc = (tid * 4) & 31;   // 0..28 step 4

    v8f acc[2][4];
    #pragma unroll
    for (int i = 0; i < 2; ++i)
        #pragma unroll
        for (int j = 0; j < 4; ++j)
            #pragma unroll
            for (int e = 0; e < 8; ++e) acc[i][j][e] = 0.0f;

    // register staging buffers (double buffer vs LDS)
    float4 aReg[4];
    uint2  bhiReg[4], bloReg[4];

    auto loadTile = [&](int k0) {
        #pragma unroll
        for (int i = 0; i < 4; ++i) {
            const int r = r0 + 32 * i;
            aReg[i] = *reinterpret_cast<const float4*>(
                &A[(size_t)(m0 + r) * lda + k0 + kc]);
        }
        #pragma unroll
        for (int i = 0; i < 4; ++i) {
            const int r = r0 + 32 * i;
            const size_t g = (size_t)(n0 + r) * ldb + k0 + kc;
            bhiReg[i] = *reinterpret_cast<const uint2*>(&Bhi[g]);
            bloReg[i] = *reinterpret_cast<const uint2*>(&Blo[g]);
        }
    };

    auto storeTile = [&]() {
        #pragma unroll
        for (int i = 0; i < 4; ++i) {
            const int r = r0 + 32 * i;
            const float f[4] = {aReg[i].x, aReg[i].y, aReg[i].z, aReg[i].w};
            Pack4 hi, lo;
            #pragma unroll
            for (int j = 0; j < 4; ++j) {
                const __bf16 h = (__bf16)f[j];
                hi.b[j] = h;
                lo.b[j] = (__bf16)(f[j] - (float)h);
            }
            *reinterpret_cast<uint2*>(&sAhi[r * LDSROW + kc]) = hi.u;
            *reinterpret_cast<uint2*>(&sAlo[r * LDSROW + kc]) = lo.u;
        }
        #pragma unroll
        for (int i = 0; i < 4; ++i) {
            const int r = r0 + 32 * i;
            *reinterpret_cast<uint2*>(&sBhi[r * LDSROW + kc]) = bhiReg[i];
            *reinterpret_cast<uint2*>(&sBlo[r * LDSROW + kc]) = bloReg[i];
        }
    };

    loadTile(0);

    for (int k0 = 0; k0 < K; k0 += BK) {
        storeTile();
        __syncthreads();
        // issue next tile's global loads now; latency hidden by WMMA phase
        if (k0 + BK < K) loadTile(k0 + BK);

        // ---- compute: 24 WMMAs / wave / K-step (hi*hi + hi*lo + lo*hi) ----
        v16bf bhi[4], blo[4];
        #pragma unroll
        for (int nt = 0; nt < 4; ++nt) {
            bhi[nt] = ldsFrag(&sBhi[(wn * 64 + nt * 16) * LDSROW], lane);
            blo[nt] = ldsFrag(&sBlo[(wn * 64 + nt * 16) * LDSROW], lane);
        }
        #pragma unroll
        for (int mt = 0; mt < 2; ++mt) {
            const v16bf ahi = ldsFrag(&sAhi[(wm * 32 + mt * 16) * LDSROW], lane);
            const v16bf alo = ldsFrag(&sAlo[(wm * 32 + mt * 16) * LDSROW], lane);
            #pragma unroll
            for (int nt = 0; nt < 4; ++nt) {
                acc[mt][nt] = __builtin_amdgcn_wmma_f32_16x16x32_bf16(
                    false, ahi, false, bhi[nt], (short)0, acc[mt][nt], false, false);
                acc[mt][nt] = __builtin_amdgcn_wmma_f32_16x16x32_bf16(
                    false, ahi, false, blo[nt], (short)0, acc[mt][nt], false, false);
                acc[mt][nt] = __builtin_amdgcn_wmma_f32_16x16x32_bf16(
                    false, alo, false, bhi[nt], (short)0, acc[mt][nt], false, false);
            }
        }
        __syncthreads();
    }

    // ---- epilogue: bias (+ReLU); C/D layout: n = lane&15, m = r + 8*(lane>=16)
    const int nlane = lane & 15;
    const int mhalf = (lane >> 4) * 8;
    #pragma unroll
    for (int mt = 0; mt < 2; ++mt) {
        #pragma unroll
        for (int nt = 0; nt < 4; ++nt) {
            const int n  = n0 + wn * 64 + nt * 16 + nlane;
            const float bv = bias[n];
            #pragma unroll
            for (int r = 0; r < 8; ++r) {
                const int m = m0 + wm * 32 + mt * 16 + mhalf + r;
                float v = acc[mt][nt][r] + bv;
                if (doRelu) v = fmaxf(v, 0.0f);
                C[(size_t)m * ldc + n] = v;
            }
        }
    }
}

extern "C" void kernel_launch(void* const* d_in, const int* in_sizes, int n_in,
                              void* d_out, int out_size, void* d_ws, size_t ws_size,
                              hipStream_t stream) {
    (void)in_sizes; (void)n_in; (void)out_size; (void)ws_size;
    const float* z2 = (const float*)d_in[1];        // [32768, 2048]
    const float* cb = (const float*)d_in[2];        // [16, 512]
    const float* W1 = (const float*)d_in[5];        // [1024, 2560]
    const float* b1 = (const float*)d_in[6];        // [1024]
    const float* W2 = (const float*)d_in[7];        // [1024, 1024]
    const float* b2 = (const float*)d_in[8];        // [1024]
    float* out = (float*)d_out;                     // [32768, 1024]

    const int Bm = 32768, L = 1024, S = 512;
    const int K1 = 2 * L;                           // 2048
    const int K2 = L;                               // 1024

    // workspace layout
    char* ws = (char*)d_ws;
    float*  b1p  = (float*)ws;                      ws += 4096;
    __bf16* w1hi = (__bf16*)ws;                     ws += (size_t)L * K1 * 2;
    __bf16* w1lo = (__bf16*)ws;                     ws += (size_t)L * K1 * 2;
    __bf16* w2hi = (__bf16*)ws;                     ws += (size_t)L * K2 * 2;
    __bf16* w2lo = (__bf16*)ws;                     ws += (size_t)L * K2 * 2;
    float*  h    = (float*)ws;                      // [32768, 1024], L2-resident

    prep_bias<<<1, 1024, 0, stream>>>(cb, W1, b1, b1p);
    split_w<<<256, 256, 0, stream>>>(W1, 2 * L + S, K1, L, w1hi, w1lo);
    split_w<<<256, 256, 0, stream>>>(W2, K2, K2, L, w2hi, w2lo);

    dim3 grid(Bm / BM, L / BN);
    // GEMM1: h = relu(z2 @ W1a^T + b1')
    gemm_bf16x2<<<grid, 256, 0, stream>>>(z2, K1, w1hi, w1lo, K1, b1p, h, L, K1, 1);
    // GEMM2: out = h @ W2^T + b2
    gemm_bf16x2<<<grid, 256, 0, stream>>>(h, L, w2hi, w2lo, L, b2, out, L, K2, 0);
}
